// DPNModel_7653631722144
// MI455X (gfx1250) — matrix operand order repaired and checked by reading, not compiled, not run on hardware
//
#include <hip/hip_runtime.h>
#include <hip/hip_bf16.h>

typedef __attribute__((ext_vector_type(16))) _Float16 v16h;
typedef __attribute__((ext_vector_type(8)))  float    v8f;
typedef __attribute__((ext_vector_type(4)))  unsigned v4u;
typedef __attribute__((ext_vector_type(8)))  int      v8i;
typedef __attribute__((ext_vector_type(4)))  int      v4i;

#if defined(__HIP_DEVICE_COMPILE__) && \
    __has_builtin(__builtin_amdgcn_tensor_load_to_lds) && \
    __has_builtin(__builtin_amdgcn_s_wait_tensorcnt)
#define USE_TDM 1
#else
#define USE_TDM 0
#endif

#if USE_TDM
// ---------------------------------------------------------------------------
// TDM 2D tile load: 16 rows x 64 cols of fp32, row stride = strideElems,
// zero-fill outside (remK x remN). D# bit layout per CDNA5 ISA ch.8.
// ---------------------------------------------------------------------------
__device__ __forceinline__ void tdm_load_tile2d(unsigned lds_off, const float* gptr,
                                                int remK, int remN, int strideElems) {
    unsigned long long ga = (unsigned long long)gptr;
    v4u g0;
    g0[0] = 1u;                                           // count=1, user mode
    g0[1] = lds_off;                                      // lds_addr (bytes)
    g0[2] = (unsigned)ga;                                 // global_addr[31:0]
    g0[3] = (unsigned)((ga >> 32) & 0x01ffffffull)        // global_addr[56:32]
          | 0x80000000u;                                  // type=2 (image)
    unsigned td0 = (unsigned)(remK < 0 ? 0 : remK);       // tensor_dim0 (K left)
    unsigned td1 = (unsigned)(remN < 0 ? 0 : remN);       // tensor_dim1 (rows left)
    unsigned long long st = (unsigned long long)(unsigned)strideElems;
    v8i g1;
    g1[0] = (int)(2u << 16);                              // data_size=4B; mask=0
    g1[1] = (int)((td0 & 0xffffu) << 16);                 // dim0[15:0]
    g1[2] = (int)((td0 >> 16) | ((td1 & 0xffffu) << 16)); // dim0[31:16] | dim1[15:0]
    g1[3] = (int)((td1 >> 16) | (64u << 16));             // dim1[31:16] | tile_dim0=64
    g1[4] = 16;                                           // tile_dim1=16, tile_dim2=0
    g1[5] = (int)(unsigned)(st & 0xffffffffull);          // dim0_stride[31:0]
    g1[6] = (int)(unsigned)((st >> 32) & 0xffffull);      // dim0_stride[47:32]
    g1[7] = 0;                                            // dim1_stride=0
    v4i z4 = {0, 0, 0, 0};
#if __has_include(<hip/amd_detail/amd_gfx1250_TDM.h>)
    v8i z8 = {0, 0, 0, 0, 0, 0, 0, 0};
    __builtin_amdgcn_tensor_load_to_lds(g0, g1, z4, z4, z8, 0);
#else
    __builtin_amdgcn_tensor_load_to_lds(g0, g1, z4, z4, 0);
#endif
}
#endif

// ---------------------------------------------------------------------------
// Generic implicit-GEMM conv (stride 1, zero pad, rhs dilation) via WMMA.
//   out[co, oy, ox] = act( bias[co] + sum_{ci,kh,kw} W[co,ci,kh,kw] *
//                          in[ci, oy-pad+dil*kh, ox-pad+dil*kw] )
// GEMM mapping: M = Hout*Wout, N = Cout, K = Cin*k*k, chunked 64-wide in K
// (two v_wmma_f32_16x16x32_f16 per chunk). Block = 128 threads = 4 waves;
// waves share the A tile (im2col gather, f16 in LDS) and each wave owns a
// 16-wide N slice whose fp32 weight tile is DMA'd by the Tensor Data Mover
// straight into LDS (s_wait_tensorcnt to consume). Weights cross HBM once
// as fp32 (streaming them is the 23.3 TB/s roofline limiter) and are
// converted to f16 during fragment load.
// ---------------------------------------------------------------------------
__global__ __launch_bounds__(128)
void conv_wmma_kernel(const float* __restrict__ in, const float* __restrict__ wt,
                      const float* __restrict__ bias, float* __restrict__ out,
                      int Cin, int Hin, int Win, int Cout, int Hout, int Wout,
                      int kk, int dil, int pad, int act) {
    const int M    = Hout * Wout;
    const int KK   = kk * kk;
    const int Ktot = Cin * KK;
    const int lane = threadIdx.x & 31;
    const int wv   = threadIdx.x >> 5;
    const int g    = lane >> 4;      // half-wave group
    const int r    = lane & 15;
    const int m0   = blockIdx.x * 16;
    const int n0   = (blockIdx.y * 4 + wv) * 16;

    __shared__ _Float16 As[16 * 64];      // A tile: [m][k]   (f16)
    __shared__ float    Bs[4][16 * 64];   // per-wave B tile: [n][k] (fp32)

    v8f acc = {};

    for (int kb = 0; kb < Ktot; kb += 64) {
        __syncthreads();

#if USE_TDM
        // Async DMA of this wave's 16(co) x 64(k) weight tile into LDS.
        tdm_load_tile2d((unsigned)(unsigned long long)&Bs[wv][0],
                        wt + (long)n0 * Ktot + kb,
                        Ktot - kb, Cout - n0, Ktot);
#else
        // Fallback: manual staging of the weight tile.
        for (int i = lane; i < 1024; i += 32) {
            int row = i >> 6, col = i & 63;
            int kg = kb + col, co = n0 + row;
            Bs[wv][i] = (kg < Ktot && co < Cout) ? wt[(long)co * Ktot + kg] : 0.0f;
        }
#endif

        // ---- Stage A: 16(m) x 64(k) im2col patch, gathered on the fly ----
        for (int i = threadIdx.x; i < 1024; i += 128) {
            int m = i >> 6, kloc = i & 63;
            int kg = kb + kloc;
            int mg = m0 + m;
            float v = 0.0f;
            if (kg < Ktot && mg < M) {
                int ci  = kg / KK;
                int rem = kg - ci * KK;
                int kh  = rem / kk, kw = rem - kh * kk;
                int oy  = mg / Wout, ox = mg - oy * Wout;
                int iy  = oy - pad + dil * kh;
                int ix  = ox - pad + dil * kw;
                if (iy >= 0 && iy < Hin && ix >= 0 && ix < Win)
                    v = in[(long)ci * Hin * Win + (long)iy * Win + ix];
            }
            As[i] = (_Float16)v;
        }
        __syncthreads();
#if USE_TDM
        __builtin_amdgcn_s_wait_tensorcnt(0);   // wave-private B tile ready
#endif

        // ---- Fragments per ISA 7.12.2 layouts; two WMMAs per 64-wide chunk ----
        v16h a0, a1, b0, b1;
        #pragma unroll
        for (int e = 0; e < 16; ++e) {
            int ka = e + 8 * g + ((e >= 8) ? 8 : 0);   // A: 16-bit 16x32 layout
            a0[e] = As[r * 64 + ka];
            a1[e] = As[r * 64 + 32 + ka];
            int kbx = e + 8 * g + ((e >= 8) ? 8 : 0);  // B: same K order as A, N=lane
            b0[e] = (_Float16)Bs[wv][r * 64 + kbx];
            b1[e] = (_Float16)Bs[wv][r * 64 + 32 + kbx];
        }
        acc = __builtin_amdgcn_wmma_f32_16x16x32_f16(
                  false, a0, false, b0, (short)0, acc, false, false);
        acc = __builtin_amdgcn_wmma_f32_16x16x32_f16(
                  false, a1, false, b1, (short)0, acc, false, false);
    }

    // ---- C/D layout: lane -> N = lane&15, VGPR v -> M = v + 8*(lane>>4) ----
    #pragma unroll
    for (int v = 0; v < 8; ++v) {
        int mg = m0 + v + 8 * g;
        int co = n0 + r;
        if (mg < M && co < Cout) {
            float val = acc[v] + bias[co];
            if (act) val = (val >= 0.0f) ? val : 0.01f * val;   // LeakyReLU
            out[(long)co * M + mg] = val;
        }
    }
}

// ---------------------------------------------------------------------------
// 2x2 max pool, stride 2 (VALID).
// ---------------------------------------------------------------------------
__global__ void maxpool2_kernel(const float* __restrict__ in, float* __restrict__ out,
                                int C, int Hin, int Win) {
    int Hout = Hin >> 1, Wout = Win >> 1;
    int total = C * Hout * Wout;
    int idx = blockIdx.x * blockDim.x + threadIdx.x;
    if (idx >= total) return;
    int x = idx % Wout;
    int y = (idx / Wout) % Hout;
    int c = idx / (Wout * Hout);
    const float* p = in + (long)c * Hin * Win + (long)(2 * y) * Win + 2 * x;
    out[idx] = fmaxf(fmaxf(p[0], p[1]), fmaxf(p[Win], p[Win + 1]));
}

// ---------------------------------------------------------------------------
// Bilinear upsample (align_corners=True) 18x18 -> 128x128, then sigmoid.
// ---------------------------------------------------------------------------
__global__ void upsample_sigmoid_kernel(const float* __restrict__ in,
                                        float* __restrict__ out) {
    int idx = blockIdx.x * blockDim.x + threadIdx.x;
    if (idx >= 21 * 128 * 128) return;
    int x = idx & 127;
    int y = (idx >> 7) & 127;
    int c = idx >> 14;
    const float s = 17.0f / 127.0f;        // (in-1)/(out-1)
    float fy = y * s, fx = x * s;
    int y0 = (int)floorf(fy); int y1 = (y0 + 1 > 17) ? 17 : y0 + 1; float ry = fy - y0;
    int x0 = (int)floorf(fx); int x1 = (x0 + 1 > 17) ? 17 : x0 + 1; float rx = fx - x0;
    const float* p = in + (long)c * 324;
    float v = p[y0 * 18 + x0] * (1.0f - ry) * (1.0f - rx)
            + p[y1 * 18 + x0] * ry * (1.0f - rx)
            + p[y0 * 18 + x1] * (1.0f - ry) * rx
            + p[y1 * 18 + x1] * ry * rx;
    out[idx] = 1.0f / (1.0f + expf(-v));
}

// ---------------------------------------------------------------------------
// Locally-connected 50x50 conv modulated by color distance.
// One block per pixel; 256 threads stride the 2500 taps; 21-channel partial
// sums in registers, reduced across the block with LDS float atomics.
// OOB taps contribute zero (zero-padded unary).
// ---------------------------------------------------------------------------
__global__ __launch_bounds__(256)
void local_conv_kernel(const float* __restrict__ x, const float* __restrict__ unary,
                       const float* __restrict__ lw, const float* __restrict__ lb,
                       float* __restrict__ out) {
    const int pix = blockIdx.x;          // 0 .. 16383
    const int h = pix >> 7, w = pix & 127;
    __shared__ float acc[21];
    if (threadIdx.x < 21) acc[threadIdx.x] = 0.0f;
    __syncthreads();

    const float x0 = x[pix];
    const float x1 = x[16384 + pix];
    const float x2 = x[2 * 16384 + pix];

    float p[21];
    #pragma unroll
    for (int c = 0; c < 21; ++c) p[c] = 0.0f;

    for (int t = threadIdx.x; t < 2500; t += 256) {
        int kh = t / 50, kw = t - kh * 50;
        int iy = h + kh - 25, ix = w + kw - 25;
        if (iy < 0 || iy >= 128 || ix < 0 || ix >= 128) continue;   // padded: u==0
        int q = iy * 128 + ix;
        float d0 = x0 - x[q];
        float d1 = x1 - x[16384 + q];
        float d2 = x2 - x[2 * 16384 + q];
        float sc = (d0 * d0 + d1 * d1 + d2 * d2) * lw[(size_t)pix * 2500 + t];
        #pragma unroll
        for (int c = 0; c < 21; ++c)
            p[c] += unary[(long)c * 16384 + q] * sc;
    }
    #pragma unroll
    for (int c = 0; c < 21; ++c) atomicAdd(&acc[c], p[c]);
    __syncthreads();
    if (threadIdx.x < 21)
        out[(long)threadIdx.x * 16384 + pix] = acc[threadIdx.x] + lb[threadIdx.x];
}

// ---------------------------------------------------------------------------
// smooth[c] = min_m g[m*21+c]; out = softmax(log(unary) - smooth) over c.
// ---------------------------------------------------------------------------
__global__ void min_softmax_kernel(const float* __restrict__ g2,
                                   const float* __restrict__ unary,
                                   float* __restrict__ out) {
    int pix = blockIdx.x * blockDim.x + threadIdx.x;
    if (pix >= 16384) return;
    float logit[21];
    float mx = -3.4e38f;
    #pragma unroll
    for (int c = 0; c < 21; ++c) {
        float sm = g2[(long)c * 16384 + pix];
        #pragma unroll
        for (int m = 1; m < 5; ++m)
            sm = fminf(sm, g2[(long)(m * 21 + c) * 16384 + pix]);
        float l = logf(unary[(long)c * 16384 + pix]) - sm;
        logit[c] = l;
        mx = fmaxf(mx, l);
    }
    float sum = 0.0f;
    #pragma unroll
    for (int c = 0; c < 21; ++c) { float e = expf(logit[c] - mx); logit[c] = e; sum += e; }
    float inv = 1.0f / sum;
    #pragma unroll
    for (int c = 0; c < 21; ++c) out[(long)c * 16384 + pix] = logit[c] * inv;
}

// ---------------------------------------------------------------------------
// Host orchestration (graph-capture safe: launches only).
// ---------------------------------------------------------------------------
struct ConvSpec { int cin, cout, k, dil, pad; };

extern "C" void kernel_launch(void* const* d_in, const int* in_sizes, int n_in,
                              void* d_out, int out_size, void* d_ws, size_t ws_size,
                              hipStream_t stream) {
    (void)in_sizes; (void)n_in; (void)out_size; (void)ws_size;

    static const ConvSpec specs[16] = {
        {3, 64, 3, 1, 1},   {64, 64, 3, 1, 1},
        {64, 128, 3, 1, 1}, {128, 128, 3, 1, 1},
        {128, 256, 3, 1, 1},{256, 256, 3, 1, 1},{256, 256, 3, 1, 1},
        {256, 512, 3, 1, 1},{512, 512, 3, 1, 1},{512, 512, 3, 1, 1},
        {512, 512, 3, 2, 2},{512, 512, 3, 2, 2},{512, 512, 3, 2, 2},
        {512, 4096, 7, 4, 12},
        {4096, 4096, 1, 1, 1},
        {4096, 21, 1, 1, 0}};

    // Workspace layout (floats). Peak activation = 4096*18*18 = 1,327,104.
    float* ws    = (float*)d_ws;
    float* ping  = ws;
    float* pong  = ping  + 1327104;
    float* unary = pong  + 1327104;                 // 21*128*128
    float* gbuf  = unary + 344064;                  // 105*128*128
    float* g2buf = gbuf  + 1720320;                 // 105*128*128

    auto conv = [&](const float* in, int Cin, int Hin, int Win,
                    const float* w, const float* b, int Cout,
                    int k, int dil, int pad, float* outp, int act,
                    int& Hout, int& Wout) {
        int eff = dil * (k - 1) + 1;
        Hout = Hin + 2 * pad - eff + 1;
        Wout = Win + 2 * pad - eff + 1;
        int M = Hout * Wout;
        dim3 grid((M + 15) / 16, (Cout + 63) / 64);
        conv_wmma_kernel<<<grid, 128, 0, stream>>>(in, w, b, outp,
            Cin, Hin, Win, Cout, Hout, Wout, k, dil, pad, act);
    };

    // ---- dilated-VGG backbone (leaky after every conv; pools after 1,3,6) ----
    const float* cur = (const float*)d_in[0];
    int C = 3, H = 128, W = 128;
    float* bufs[2] = {ping, pong};
    int nxt = 0;
    for (int i = 0; i < 16; ++i) {
        const float* wsrc = (const float*)d_in[1 + 2 * i];
        const float* bsrc = (const float*)d_in[2 + 2 * i];
        int Ho, Wo;
        float* dst = bufs[nxt];
        conv(cur, C, H, W, wsrc, bsrc, specs[i].cout,
             specs[i].k, specs[i].dil, specs[i].pad, dst, /*leaky=*/1, Ho, Wo);
        cur = dst; C = specs[i].cout; H = Ho; W = Wo; nxt ^= 1;
        if (i == 1 || i == 3 || i == 6) {
            float* pdst = bufs[nxt];
            int total = C * (H >> 1) * (W >> 1);
            maxpool2_kernel<<<(total + 255) / 256, 256, 0, stream>>>(cur, pdst, C, H, W);
            H >>= 1; W >>= 1;
            cur = pdst; nxt ^= 1;
        }
    }
    // cur = backbone output: 21 x 18 x 18

    // ---- unary = sigmoid(upsample_bilinear_ac(h, 128, 128)) ----
    upsample_sigmoid_kernel<<<(21 * 16384 + 255) / 256, 256, 0, stream>>>(cur, unary);

    // ---- locally-connected 50x50 color-modulated conv (+local_b) ----
    const float* local_w = (const float*)d_in[33];
    const float* local_b = (const float*)d_in[34];
    float* local_out = ping;
    local_conv_kernel<<<16384, 256, 0, stream>>>((const float*)d_in[0], unary,
                                                 local_w, local_b, local_out);

    // ---- lin1 (channel linear == 1x1 conv, 21->21) ----
    int Ho, Wo;
    conv(local_out, 21, 128, 128, (const float*)d_in[35], (const float*)d_in[36],
         21, 1, 1, 0, pong, /*act=*/0, Ho, Wo);

    // ---- global 9x9 conv 21 -> 105 ----
    conv(pong, 21, 128, 128, (const float*)d_in[37], (const float*)d_in[38],
         105, 9, 1, 4, gbuf, /*act=*/0, Ho, Wo);

    // ---- lin2 (1x1 conv, 105->105) ----
    conv(gbuf, 105, 128, 128, (const float*)d_in[39], (const float*)d_in[40],
         105, 1, 1, 0, g2buf, /*act=*/0, Ho, Wo);

    // ---- min over mixture dim + softmax(log(unary) - smooth) ----
    min_softmax_kernel<<<(16384 + 255) / 256, 256, 0, stream>>>(g2buf, unary,
                                                                (float*)d_out);
}
